// DenseFullAttention_27204322853611
// MI455X (gfx1250) — matrix operand order repaired and checked by reading, not compiled, not run on hardware
//
#include <hip/hip_runtime.h>
#include <cstdint>

// Problem constants
#define BB  4
#define SS  2048
#define DD  2048
#define HH  16
#define HDD 128
#define BS  (BB * SS)          // 8192 rows

typedef __attribute__((ext_vector_type(16))) __bf16        v16bf;
typedef __attribute__((ext_vector_type(8)))  float         v8f;
typedef __attribute__((ext_vector_type(4)))  unsigned int  u32x4;
typedef __attribute__((ext_vector_type(8)))  int           i32x8;
typedef __attribute__((ext_vector_type(4)))  int           i32x4;

union Frag { v16bf v; u32x4 u[2]; };   // 32 bytes: one bf16 A or B fragment per lane

__device__ __forceinline__ uint16_t f2bf(float f) {
  union { float f; uint32_t u; } c; c.f = f;
  uint32_t u = c.u;
  return (uint16_t)((u + 0x7FFFu + ((u >> 16) & 1u)) >> 16);   // round-to-nearest-even
}

// --------------------------------------------------------------- TDM helper --
// 2-D tensor tile load via the Tensor Data Mover (ISA 08_async_tensor §7/§8).
// D# group0: count=1 | lds_addr | global_addr[56:0] | type=2 ("image")
// D# group1: data_size=1(2B) | tensor_dim0/1 | tile_dim0/1 | tensor_dim0_stride
// groups 2/3 zero => 2-D tensor. Tracked by TENSORcnt.
// This toolchain exposes the 6-arg builtin:
//   (u32x4 g0, i32x8 g1, i32x4 g2, i32x4 g3, i32x8 extra, i32 cpol)
// lds_off: low 32 bits of a generic __shared__ pointer == wave-relative LDS
// byte offset (ISA §10.2: the LDS aperture keeps the low 32 address bits).
__device__ __forceinline__ void tdm_load_2d(uint32_t lds_off, uint64_t gaddr,
                                            uint32_t tile_x, uint32_t tile_y,
                                            uint32_t tensor_x, uint32_t tensor_y,
                                            uint32_t stride_x /* elements */) {
  u32x4 g0 = { 1u,                                   // count=1, user mode
               lds_off,                              // lds_addr (bytes)
               (uint32_t)gaddr,                      // global_addr[31:0]
               (uint32_t)((gaddr >> 32) & 0x1FFFFFFull) | 0x80000000u }; // [56:32] | type=2
  i32x8 g1 = { (int)(1u << 16),                                   // data_size = 2 bytes
               (int)((tensor_x & 0xFFFFu) << 16),                 // tensor_dim0[15:0]
               (int)((tensor_x >> 16) | ((tensor_y & 0xFFFFu) << 16)),
               (int)((tensor_y >> 16) | ((tile_x & 0xFFFFu) << 16)),
               (int)(tile_y & 0xFFFFu),                           // tile_dim1, tile_dim2=0
               (int)stride_x,                                     // tensor_dim0_stride[31:0]
               0, 0 };
  i32x4 gz4 = { 0, 0, 0, 0 };
  i32x8 gz8 = { 0, 0, 0, 0, 0, 0, 0, 0 };
  __builtin_amdgcn_tensor_load_to_lds(g0, g1, gz4, gz4, gz8, 0);
}

// ---------------------------------------------------------------- converts ---
__global__ __launch_bounds__(256) void conv_x_kernel(const float* __restrict__ x,
                                                     uint16_t* __restrict__ xb, int n4) {
  int i = blockIdx.x * blockDim.x + threadIdx.x;
  if (i >= n4) return;
  const float4 f = ((const float4*)x)[i];
  union { uint16_t h[4]; unsigned long long u; } p;
  p.h[0] = f2bf(f.x); p.h[1] = f2bf(f.y); p.h[2] = f2bf(f.z); p.h[3] = f2bf(f.w);
  ((unsigned long long*)xb)[i] = p.u;
}

// fp32 [K=2048][N=2048] -> bf16 transposed [N][K] (so B tiles are row-contiguous)
__global__ __launch_bounds__(256) void conv_wT_kernel(const float* __restrict__ w0, const float* __restrict__ w1,
                                                      const float* __restrict__ w2, const float* __restrict__ w3,
                                                      uint16_t* __restrict__ o0, uint16_t* __restrict__ o1,
                                                      uint16_t* __restrict__ o2, uint16_t* __restrict__ o3) {
  const int idx = blockIdx.x * blockDim.x + threadIdx.x;   // 0 .. 2048*2048-1
  const int k = idx % DD;
  const int ncol = idx / DD;
  const float* w; uint16_t* o;
  switch (blockIdx.y) {
    case 0:  w = w0; o = o0; break;
    case 1:  w = w1; o = o1; break;
    case 2:  w = w2; o = o2; break;
    default: w = w3; o = o3; break;
  }
  o[(size_t)ncol * DD + k] = f2bf(w[(size_t)k * DD + ncol]);
}

// ------------------------------------------------------------- QKV + RoPE ----
// Block = 4 waves sharing one head; each wave owns a 16-row M tile (16x128 out).
// The shared 128-col x 128-k B panel is staged into LDS by the TDM,
// double-buffered, synced with s_wait_tensorcnt + workgroup barrier.
// mode 0 = Q (RoPE, store [b,h,s,hd]), 1 = K (same), 2 = V (store [b,h,hd,s]).
__global__ __launch_bounds__(128) void qkv_rope_kernel(const uint16_t* __restrict__ xb,
                                                       const uint16_t* __restrict__ wqt,
                                                       const uint16_t* __restrict__ wkt,
                                                       const uint16_t* __restrict__ wvt,
                                                       const float* __restrict__ cosp,
                                                       const float* __restrict__ sinp,
                                                       uint16_t* __restrict__ Qb,
                                                       uint16_t* __restrict__ Kb,
                                                       uint16_t* __restrict__ Vt) {
  constexpr int KC = 128;                    // K-chunk staged per TDM transfer
  constexpr int NCHUNK = DD / KC;            // 16
  __shared__ uint16_t bpanel[2][HDD * KC];   // 2 x 32 KiB double buffer

  const int lane = threadIdx.x & 31;
  const int wave = threadIdx.x >> 5;
  const int mt   = blockIdx.x * 4 + wave;    // 0..511
  const int h    = blockIdx.y;               // head
  const int mode = blockIdx.z;               // 0=Q 1=K 2=V
  const int half = lane >> 4;
  const int n    = lane & 15;
  const int row0 = mt * 16;

  const uint16_t* wt = (mode == 0) ? wqt : (mode == 1) ? wkt : wvt;
  // byte address of this head's column panel inside the transposed weights
  const uint64_t wbase = (uint64_t)(uintptr_t)wt + (uint64_t)(h * HDD) * DD * 2u;

  v8f acc[8];
#pragma unroll
  for (int t = 0; t < 8; ++t) acc[t] = (v8f)0.0f;

  // prologue: stage chunk 0 into buffer 0
  if (wave == 0) {
    tdm_load_2d((uint32_t)(uintptr_t)&bpanel[0][0], wbase,
                /*tile  */ KC, HDD,
                /*tensor*/ DD, DD,
                /*stride*/ DD);
    __builtin_amdgcn_s_wait_tensorcnt(0);
  }
  __syncthreads();

  for (int kc = 0; kc < NCHUNK; ++kc) {
    const int cur = kc & 1;
    if (wave == 0 && kc + 1 < NCHUNK) {      // async-stage next chunk
      tdm_load_2d((uint32_t)(uintptr_t)&bpanel[cur ^ 1][0],
                  wbase + (uint64_t)(kc + 1) * KC * 2u,
                  KC, HDD, DD, DD, DD);
    }
    const uint16_t* pl = bpanel[cur];
    const uint16_t* arow = xb + (size_t)(row0 + n) * DD + kc * KC;
    if (kc + 1 < NCHUNK) __builtin_prefetch(arow + KC, 0, 0);
#pragma unroll
    for (int cc = 0; cc < KC / 32; ++cc) {
      Frag a;   // A fragment: row m = lane&15, 16 consecutive K per lane half
      a.u[0] = *(const u32x4*)(arow + cc * 32 + half * 16);
      a.u[1] = *(const u32x4*)(arow + cc * 32 + half * 16 + 8);
#pragma unroll
      for (int t = 0; t < 8; ++t) {
        Frag b;   // B fragment from LDS panel: column-local row, 16 consecutive K
        const uint16_t* bb = pl + (t * 16 + n) * KC + cc * 32 + half * 16;
        b.u[0] = *(const u32x4*)(bb);
        b.u[1] = *(const u32x4*)(bb + 8);
        acc[t] = __builtin_amdgcn_wmma_f32_16x16x32_bf16(false, a.v, false, b.v,
                                                         (short)0, acc[t], false, false);
      }
    }
    if (wave == 0 && kc + 1 < NCHUNK) __builtin_amdgcn_s_wait_tensorcnt(0);
    __syncthreads();                          // next buffer ready / current drained
  }

  const int b  = row0 / SS;
  const int s0 = row0 % SS;

  if (mode == 2) {
    // V transposed: Vt[((b*H+h)*HD + hd)*S + s]; lane holds 8 consecutive s -> 16B store
#pragma unroll
    for (int t = 0; t < 8; ++t) {
      union { uint16_t hh[8]; u32x4 v; } pk;
#pragma unroll
      for (int r = 0; r < 8; ++r) pk.hh[r] = f2bf(acc[t][r]);
      const int hd = t * 16 + n;
      uint16_t* dst = Vt + ((size_t)((b * HH + h) * HDD + hd)) * SS + s0 + half * 8;
      *(u32x4*)dst = pk.v;
    }
  } else {
    // RoPE: pair (hd, hd+64) = C tiles t and t+4, same lane/VGPR, same cos/sin[hd%64]
#pragma unroll
    for (int r = 0; r < 8; ++r) {
      const int s = s0 + half * 8 + r;
#pragma unroll
      for (int t = 0; t < 4; ++t) {
        const int hd   = t * 16 + n;          // 0..63
        const float c  = cosp[s * (HDD / 2) + hd];
        const float sn = sinp[s * (HDD / 2) + hd];
        const float lo = acc[t][r], hi = acc[t + 4][r];
        acc[t][r]     = lo * c - hi * sn;     // q[hd]*cos - q[hd+64]*sin
        acc[t + 4][r] = hi * c + lo * sn;     // q[hd+64]*cos + q[hd]*sin
      }
    }
    uint16_t* dst = (mode == 0) ? Qb : Kb;
#pragma unroll
    for (int t = 0; t < 8; ++t)
#pragma unroll
      for (int r = 0; r < 8; ++r) {
        const int s = s0 + half * 8 + r;
        dst[((size_t)((b * HH + h) * SS + s)) * HDD + t * 16 + n] = f2bf(acc[t][r]);
      }
  }
}

// ------------------------------------------------------------- attention -----
// One wave = 16 queries of one (b,h); flash-attention over 32-key blocks.
// (Per-wave causal trip counts => no block barriers; P bounce uses wave-private LDS.)
__global__ __launch_bounds__(128) void attn_kernel(const uint16_t* __restrict__ Qb,
                                                   const uint16_t* __restrict__ Kb,
                                                   const uint16_t* __restrict__ Vt,
                                                   uint16_t* __restrict__ O) {
  __shared__ uint16_t pbuf[4][16 * 32];        // per-wave P staging (C->A layout bounce)
  const int lane = threadIdx.x & 31;
  const int wave = threadIdx.x >> 5;
  const int qt   = blockIdx.x * 4 + wave;      // 0..127
  const int bh   = blockIdx.y;                 // 0..63
  const int b = bh / HH, h = bh % HH;
  const int half = lane >> 4;
  const int n    = lane & 15;
  const int q0   = qt * 16;
  const float scale = 0.08838834764831845f;    // 1/sqrt(128)

  const uint16_t* Qh = Qb + ((size_t)bh * SS) * HDD;
  const uint16_t* Kh = Kb + ((size_t)bh * SS) * HDD;
  const uint16_t* Vh = Vt + ((size_t)bh * HDD) * SS;

  // Q A-fragments, 4 chunks of K=32 across HD=128
  Frag aQ[4];
  {
    const uint16_t* base = Qh + (size_t)(q0 + n) * HDD;
#pragma unroll
    for (int c = 0; c < 4; ++c) {
      aQ[c].u[0] = *(const u32x4*)(base + c * 32 + half * 16);
      aQ[c].u[1] = *(const u32x4*)(base + c * 32 + half * 16 + 8);
    }
  }

  v8f acc[8];
#pragma unroll
  for (int t = 0; t < 8; ++t) acc[t] = (v8f)0.0f;
  float mi[8], li[8];
#pragma unroll
  for (int r = 0; r < 8; ++r) { mi[r] = -1e30f; li[r] = 0.0f; }

  const int nblocks = (q0 + 15) / 32 + 1;      // causal: only blocks touching k <= q
  for (int kb = 0; kb < nblocks; ++kb) {
    // ---- logits: S = Q K^T for 16q x 32k (two 16x16 tiles) ----
    v8f cl[2];
#pragma unroll
    for (int t2 = 0; t2 < 2; ++t2) {
      v8f c = (v8f)0.0f;
      const int key = kb * 32 + t2 * 16 + n;   // B column = key
      const uint16_t* kbase = Kh + (size_t)key * HDD + half * 16;
#pragma unroll
      for (int cc = 0; cc < 4; ++cc) {
        Frag bk;                               // B = K^T: 16 consecutive hd per lane
        bk.u[0] = *(const u32x4*)(kbase + cc * 32);
        bk.u[1] = *(const u32x4*)(kbase + cc * 32 + 8);
        c = __builtin_amdgcn_wmma_f32_16x16x32_bf16(false, aQ[cc].v, false, bk.v,
                                                    (short)0, c, false, false);
      }
      cl[t2] = c;
    }

    // ---- online softmax (fp32, wave-half shuffle reductions) ----
    float corr[8], p0[8], p1[8];
#pragma unroll
    for (int r = 0; r < 8; ++r) {
      const int q  = q0 + half * 8 + r;
      const int k0 = kb * 32 + n;
      const int k1 = kb * 32 + 16 + n;
      float v0 = (k0 <= q) ? cl[0][r] * scale : -1e30f;
      float v1 = (k1 <= q) ? cl[1][r] * scale : -1e30f;
      float rm = fmaxf(v0, v1);
      rm = fmaxf(rm, __shfl_xor(rm, 1, 32));
      rm = fmaxf(rm, __shfl_xor(rm, 2, 32));
      rm = fmaxf(rm, __shfl_xor(rm, 4, 32));
      rm = fmaxf(rm, __shfl_xor(rm, 8, 32));
      const float mn = fmaxf(mi[r], rm);
      const float cr = __expf(mi[r] - mn);
      const float e0 = __expf(v0 - mn);
      const float e1 = __expf(v1 - mn);
      float rs = e0 + e1;
      rs += __shfl_xor(rs, 1, 32);
      rs += __shfl_xor(rs, 2, 32);
      rs += __shfl_xor(rs, 4, 32);
      rs += __shfl_xor(rs, 8, 32);
      li[r] = li[r] * cr + rs;
      mi[r] = mn;
      corr[r] = cr; p0[r] = e0; p1[r] = e1;
    }
#pragma unroll
    for (int t = 0; t < 8; ++t)
#pragma unroll
      for (int r = 0; r < 8; ++r) acc[t][r] *= corr[r];

    // ---- P (C layout) -> LDS -> A fragment ----
    uint16_t* pl = pbuf[wave];
#pragma unroll
    for (int r = 0; r < 8; ++r) {
      pl[(half * 8 + r) * 32 + n]      = f2bf(p0[r]);
      pl[(half * 8 + r) * 32 + 16 + n] = f2bf(p1[r]);
    }
    asm volatile("s_wait_dscnt 0" ::: "memory");   // wave-private LDS RAW fence

    Frag aP;
    {
      const uint16_t* pb = pl + n * 32;            // A row m = lane&15
      aP.u[0] = *(const u32x4*)(pb + half * 16);
      aP.u[1] = *(const u32x4*)(pb + half * 16 + 8);
    }

    // ---- acc += P @ V  (V^T: 16 consecutive keys contiguous per lane) ----
    const uint16_t* vbase = Vh + (size_t)(kb * 32 + half * 16);
#pragma unroll
    for (int t = 0; t < 8; ++t) {
      Frag bv;
      const uint16_t* vb = vbase + (size_t)(t * 16 + n) * SS;
      bv.u[0] = *(const u32x4*)(vb);
      bv.u[1] = *(const u32x4*)(vb + 8);
      acc[t] = __builtin_amdgcn_wmma_f32_16x16x32_bf16(false, aP.v, false, bv.v,
                                                       (short)0, acc[t], false, false);
    }
  }

  // ---- normalize and store O[b, s, h*HD + hd] (bf16) ----
  float inv[8];
#pragma unroll
  for (int r = 0; r < 8; ++r) inv[r] = 1.0f / li[r];
#pragma unroll
  for (int t = 0; t < 8; ++t)
#pragma unroll
    for (int r = 0; r < 8; ++r) {
      const int s = q0 + half * 8 + r;
      O[((size_t)(b * SS + s)) * (HH * HDD) + h * HDD + t * 16 + n] = f2bf(acc[t][r] * inv[r]);
    }
}

// --------------------------------------------------------- output projection --
// Block = 4 waves sharing one 64-column N block; TDM stages the 64x128 B panel.
__global__ __launch_bounds__(128) void oproj_kernel(const uint16_t* __restrict__ O,
                                                    const uint16_t* __restrict__ wot,
                                                    float* __restrict__ out) {
  constexpr int KC = 128;
  constexpr int NCHUNK = DD / KC;            // 16
  __shared__ uint16_t bpanel[2][64 * KC];    // 2 x 16 KiB

  const int lane = threadIdx.x & 31;
  const int wave = threadIdx.x >> 5;
  const int mt   = blockIdx.x * 4 + wave;    // 0..511
  const int nb   = blockIdx.y;               // 0..31 (64 cols each)
  const int half = lane >> 4;
  const int n    = lane & 15;
  const int row0 = mt * 16;

  const uint64_t wbase = (uint64_t)(uintptr_t)wot + (uint64_t)(nb * 64) * DD * 2u;

  v8f acc[4];
#pragma unroll
  for (int t = 0; t < 4; ++t) acc[t] = (v8f)0.0f;

  if (wave == 0) {
    tdm_load_2d((uint32_t)(uintptr_t)&bpanel[0][0], wbase, KC, 64, DD, DD, DD);
    __builtin_amdgcn_s_wait_tensorcnt(0);
  }
  __syncthreads();

  for (int kc = 0; kc < NCHUNK; ++kc) {
    const int cur = kc & 1;
    if (wave == 0 && kc + 1 < NCHUNK) {
      tdm_load_2d((uint32_t)(uintptr_t)&bpanel[cur ^ 1][0],
                  wbase + (uint64_t)(kc + 1) * KC * 2u, KC, 64, DD, DD, DD);
    }
    const uint16_t* pl = bpanel[cur];
    const uint16_t* arow = O + (size_t)(row0 + n) * DD + kc * KC;
    if (kc + 1 < NCHUNK) __builtin_prefetch(arow + KC, 0, 0);
#pragma unroll
    for (int cc = 0; cc < KC / 32; ++cc) {
      Frag a;
      a.u[0] = *(const u32x4*)(arow + cc * 32 + half * 16);
      a.u[1] = *(const u32x4*)(arow + cc * 32 + half * 16 + 8);
#pragma unroll
      for (int t = 0; t < 4; ++t) {
        Frag bw;
        const uint16_t* bb = pl + (t * 16 + n) * KC + cc * 32 + half * 16;
        bw.u[0] = *(const u32x4*)(bb);
        bw.u[1] = *(const u32x4*)(bb + 8);
        acc[t] = __builtin_amdgcn_wmma_f32_16x16x32_bf16(false, a.v, false, bw.v,
                                                         (short)0, acc[t], false, false);
      }
    }
    if (wave == 0 && kc + 1 < NCHUNK) __builtin_amdgcn_s_wait_tensorcnt(0);
    __syncthreads();
  }

#pragma unroll
  for (int t = 0; t < 4; ++t)
#pragma unroll
    for (int r = 0; r < 8; ++r) {
      const int row = row0 + half * 8 + r;
      out[(size_t)row * DD + nb * 64 + t * 16 + n] = acc[t][r];
    }
}

// ------------------------------------------------------------------ launch ----
extern "C" void kernel_launch(void* const* d_in, const int* in_sizes, int n_in,
                              void* d_out, int out_size, void* d_ws, size_t ws_size,
                              hipStream_t stream) {
  (void)in_sizes; (void)n_in; (void)out_size; (void)ws_size;
  const float* x   = (const float*)d_in[0];
  const float* wq  = (const float*)d_in[1];
  const float* wk  = (const float*)d_in[2];
  const float* wvp = (const float*)d_in[3];
  const float* wo  = (const float*)d_in[4];
  const float* cs  = (const float*)d_in[5];
  const float* sn  = (const float*)d_in[6];

  // Workspace layout (bf16), ~192 MiB total:
  char* ws = (char*)d_ws;
  const size_t MB = 1024u * 1024u;
  uint16_t* xb  = (uint16_t*)(ws);               // 32 MiB  [8192 x 2048]
  uint16_t* wqt = (uint16_t*)(ws + 32  * MB);    // 8 MiB   [N x K] transposed
  uint16_t* wkt = (uint16_t*)(ws + 40  * MB);
  uint16_t* wvt = (uint16_t*)(ws + 48  * MB);
  uint16_t* wot = (uint16_t*)(ws + 56  * MB);
  uint16_t* Qb  = (uint16_t*)(ws + 64  * MB);    // 32 MiB  [B,H,S,HD]
  uint16_t* Kb  = (uint16_t*)(ws + 96  * MB);    // 32 MiB  [B,H,S,HD]
  uint16_t* Vt  = (uint16_t*)(ws + 128 * MB);    // 32 MiB  [B,H,HD,S]
  uint16_t* Ob  = (uint16_t*)(ws + 160 * MB);    // 32 MiB  [B,S,H*HD]

  conv_x_kernel<<<(BS * DD / 4 + 255) / 256, 256, 0, stream>>>(x, xb, BS * DD / 4);
  conv_wT_kernel<<<dim3((DD * DD + 255) / 256, 4), 256, 0, stream>>>(
      wq, wk, wvp, wo, wqt, wkt, wvt, wot);
  qkv_rope_kernel<<<dim3(BS / 16 / 4, HH, 3), 128, 0, stream>>>(
      xb, wqt, wkt, wvt, cs, sn, Qb, Kb, Vt);
  attn_kernel<<<dim3(SS / 16 / 4, BB * HH), 128, 0, stream>>>(Qb, Kb, Vt, Ob);
  oproj_kernel<<<dim3(BS / 16 / 4, DD / 64), 128, 0, stream>>>(Ob, wot, (float*)d_out);
}
